// OffsetHead_74071005987253
// MI455X (gfx1250) — matrix operand order repaired and checked by reading, not compiled, not run on hardware
//
#include <hip/hip_runtime.h>
#include <stdint.h>

// ---------------------------------------------------------------------------
// MI455X / gfx1250 implementation.
//  - Decoder MLP fully on V_WMMA_F32_16X16X32_F16 (one 16-row tile per wave32):
//      layer1 64->32 : 4 WMMA   layer2 32->16 : 1 WMMA   layer3 16->3 : 1 WMMA (padded)
//  - Revoxelize via hash-table unique + deterministic prefix-sum ranking
// ---------------------------------------------------------------------------

typedef __attribute__((ext_vector_type(16))) _Float16 v16h;
typedef __attribute__((ext_vector_type(8)))  float    v8f;

#define EMPTY_KEY 0xFFFFFFFFFFFFFFFFull
#define F32_BIG   3.402823466e+38f

__device__ __forceinline__ uint64_t mix64(uint64_t x) {
    x ^= x >> 33; x *= 0xff51afd7ed558ccdull;
    x ^= x >> 33; x *= 0xc4ceb9fe1a85ec53ull;
    x ^= x >> 33; return x;
}

__device__ __forceinline__ int imin_(int a, int b) { return a < b ? a : b; }

// single-instruction ReLU: v_med3_f32(x, 0, +MAX)
__device__ __forceinline__ float relu1(float x) {
    return __builtin_amdgcn_fmed3f(x, 0.0f, F32_BIG);
}

// ---------------------------------------------------------------------------
// Weight conversion f32 -> f16 (tiny, once per launch)
// ---------------------------------------------------------------------------
__global__ void k_cvt_weights(const float* __restrict__ W1, const float* __restrict__ W2,
                              const float* __restrict__ W3,
                              _Float16* __restrict__ W1h, _Float16* __restrict__ W2h,
                              _Float16* __restrict__ W3h) {
    int i = blockIdx.x * blockDim.x + threadIdx.x;
    if (i < 64 * 32) W1h[i] = (_Float16)W1[i];
    if (i < 32 * 16) W2h[i] = (_Float16)W2[i];
    if (i < 16 * 3)  W3h[i] = (_Float16)W3[i];
}

// ---------------------------------------------------------------------------
// Decoder: per-wave 16-row tile, WMMA f16 in / f32 accumulate.
// ---------------------------------------------------------------------------
__global__ void __launch_bounds__(256)
k_decoder(const float* __restrict__ feats,
          const int* __restrict__ coords,
          const unsigned char* __restrict__ mask,
          const _Float16* __restrict__ W1h,
          const float* __restrict__ b1,
          const _Float16* __restrict__ W2h,
          const float* __restrict__ b2,
          const _Float16* __restrict__ W3h,
          const float* __restrict__ b3,
          float* __restrict__ off_out,
          int* __restrict__ newc,
          int N)
{
    __shared__ float h1[8][16][33];   // per-wave slice: layer1 activations
    __shared__ float h2[8][16][17];   // per-wave slice: layer2 activations
    __shared__ float h3[8][16][4];    // per-wave slice: layer3 outputs (3 cols)

    const int lane = threadIdx.x & 31;
    const int w    = threadIdx.x >> 5;          // wave in block (wave32)
    const int m    = lane & 15;                 // row within tile / column id
    const int g    = lane >> 4;                 // K-half selector
    const int wave   = blockIdx.x * (blockDim.x >> 5) + w;
    const int nwaves = gridDim.x * (blockDim.x >> 5);
    const int ntiles = (N + 15) >> 4;

    // ---- B fragments (ISA 16-bit B layout: lane holds col m, K = g*8+0..7 and 16+g*8+0..7),
    //      invariant across the whole tile loop -> load once.
    v16h B1[2][2];                                  // [kt][nt] for 64x32 weight
    for (int kt = 0; kt < 2; ++kt)
        for (int nt = 0; nt < 2; ++nt) {
            const int k0  = kt * 32 + g * 8;
            const int col = nt * 16 + m;
            v16h b;
#pragma unroll
            for (int j = 0; j < 16; ++j) b[j] = W1h[(k0 + j + (j >= 8 ? 8 : 0)) * 32 + col];
            B1[kt][nt] = b;
        }
    v16h B2;                                        // 32x16 weight
    {
        const int k0 = g * 8;
#pragma unroll
        for (int j = 0; j < 16; ++j) B2[j] = W2h[(k0 + j + (j >= 8 ? 8 : 0)) * 16 + m];
    }
    v16h B3 = {};                                   // 16x3 weight, zero-padded to 16x16/K32
    if (m < 3) {                                    // K rows 0..15 valid, cols 0..2 valid
#pragma unroll
        for (int j = 0; j < 8; ++j) B3[j] = W3h[(g * 8 + j) * 3 + m];
    }
    const float bias1[2] = { b1[m], b1[16 + m] };   // C layout: col depends only on m
    const float bias2    = b2[m];
    const float bias3    = (m < 3) ? b3[m] : 0.0f;

    for (int t = wave; t < ntiles; t += nwaves) {
        const int base = t << 4;
        const int row  = base + m;
        const int rc   = imin_(row, N - 1);         // clamp: keep EXEC all-ones for WMMA
        const float* fp = feats + (size_t)rc * 64;

        // Prefetch the next tile this wave will touch.
        {
            const int nr = imin_(base + nwaves * 16 + m, N - 1);
            __builtin_prefetch(feats + (size_t)nr * 64, 0, 1);
        }

        // ---- layer 1 --------------------------------------------------------
        v8f acc[2];
#pragma unroll
        for (int nt = 0; nt < 2; ++nt) {
            v8f c;
#pragma unroll
            for (int r = 0; r < 8; ++r) c[r] = bias1[nt];
            acc[nt] = c;
        }
#pragma unroll
        for (int kt = 0; kt < 2; ++kt) {
            // A fragment (16-bit A 16x32 layout): lane m holds K = g*8..g*8+7 and 16+g*8..
            const float4* p0 = (const float4*)(fp + kt * 32 + g * 8);
            const float4* p1 = (const float4*)(fp + kt * 32 + 16 + g * 8);
            const float4 f0 = p0[0], f1 = p0[1], f2 = p1[0], f3 = p1[1];
            v16h a;
            a[0]  = (_Float16)f0.x; a[1]  = (_Float16)f0.y; a[2]  = (_Float16)f0.z; a[3]  = (_Float16)f0.w;
            a[4]  = (_Float16)f1.x; a[5]  = (_Float16)f1.y; a[6]  = (_Float16)f1.z; a[7]  = (_Float16)f1.w;
            a[8]  = (_Float16)f2.x; a[9]  = (_Float16)f2.y; a[10] = (_Float16)f2.z; a[11] = (_Float16)f2.w;
            a[12] = (_Float16)f3.x; a[13] = (_Float16)f3.y; a[14] = (_Float16)f3.z; a[15] = (_Float16)f3.w;
#pragma unroll
            for (int nt = 0; nt < 2; ++nt)
                acc[nt] = __builtin_amdgcn_wmma_f32_16x16x32_f16(
                    false, a, false, B1[kt][nt], (short)0, acc[nt], false, false);
        }
        // ReLU + stash in LDS in C layout (row = g*8+r, col = nt*16+m)
#pragma unroll
        for (int nt = 0; nt < 2; ++nt)
#pragma unroll
            for (int r = 0; r < 8; ++r)
                h1[w][g * 8 + r][nt * 16 + m] = relu1(acc[nt][r]);

        // ---- layer 2 (intra-wave LDS transpose C-layout -> A-layout) -------
        v16h a2;
#pragma unroll
        for (int j = 0; j < 8; ++j) a2[j]     = (_Float16)h1[w][m][g * 8 + j];
#pragma unroll
        for (int j = 0; j < 8; ++j) a2[8 + j] = (_Float16)h1[w][m][16 + g * 8 + j];
        v8f c2;
#pragma unroll
        for (int r = 0; r < 8; ++r) c2[r] = bias2;
        c2 = __builtin_amdgcn_wmma_f32_16x16x32_f16(
            false, a2, false, B2, (short)0, c2, false, false);
#pragma unroll
        for (int r = 0; r < 8; ++r)
            h2[w][g * 8 + r][m] = relu1(c2[r]);

        // ---- layer 3 (16 -> 3) as zero-padded WMMA --------------------------
        v16h a3 = {};                               // K = 16..31 stays zero
#pragma unroll
        for (int j = 0; j < 8; ++j) a3[j] = (_Float16)h2[w][m][g * 8 + j];
        v8f c3;
#pragma unroll
        for (int r = 0; r < 8; ++r) c3[r] = bias3;
        c3 = __builtin_amdgcn_wmma_f32_16x16x32_f16(
            false, a3, false, B3, (short)0, c3, false, false);
        if (m < 3) {                                // scatter the 3 live columns
#pragma unroll
            for (int r = 0; r < 8; ++r) h3[w][g * 8 + r][m] = c3[r];
        }

        // ---- mask + round + store + new coords ------------------------------
        if (lane < 16 && row < N) {
            const float mk = mask[row] ? 1.0f : 0.0f;
            const float o0 = h3[w][m][0] * mk;
            const float o1 = h3[w][m][1] * mk;
            const float o2 = h3[w][m][2] * mk;
            off_out[(size_t)row * 3 + 0] = o0;
            off_out[(size_t)row * 3 + 1] = o1;
            off_out[(size_t)row * 3 + 2] = o2;
            const int4 cc = ((const int4*)coords)[row];
            int4 nc;
            nc.x = cc.x;
            nc.y = cc.y + (int)rintf(o0);           // RNE, matches jnp.round
            nc.z = cc.z + (int)rintf(o1);
            nc.w = cc.w + (int)rintf(o2);
            ((int4*)newc)[row] = nc;
        }
    }
}

// ---------------------------------------------------------------------------
// Hash-table unique over int4 rows
// ---------------------------------------------------------------------------
__global__ void k_init_table(unsigned long long* __restrict__ keys, int* __restrict__ rep,
                             int* __restrict__ cnt, float* __restrict__ sum, int cap) {
    int i = blockIdx.x * blockDim.x + threadIdx.x;
    if (i < cap) {
        keys[i] = EMPTY_KEY;
        rep[i]  = 0x7fffffff;
        cnt[i]  = 0;
        float4 z = make_float4(0.f, 0.f, 0.f, 0.f);
        ((float4*)sum)[i] = z;
    }
}

__global__ void k_hash_insert(const int* __restrict__ newc,
                              unsigned long long* __restrict__ keys,
                              int* __restrict__ rep, int* __restrict__ cnt,
                              float* __restrict__ sum, int* __restrict__ slot_of_row,
                              int N, unsigned int capm1) {
    int i = blockIdx.x * blockDim.x + threadIdx.x;
    if (i >= N) return;
    const int4 c = ((const int4*)newc)[i];
    const uint64_t key = ((uint64_t)((unsigned)(c.x + 1024) & 0xFFFFu) << 48) |
                         ((uint64_t)((unsigned)(c.y + 1024) & 0xFFFFu) << 32) |
                         ((uint64_t)((unsigned)(c.z + 1024) & 0xFFFFu) << 16) |
                         ((uint64_t)((unsigned)(c.w + 1024) & 0xFFFFu));
    unsigned idx = (unsigned)mix64(key) & capm1;
    for (;;) {
        unsigned long long prev = atomicCAS(&keys[idx], EMPTY_KEY, key);
        if (prev == EMPTY_KEY || prev == key) break;
        idx = (idx + 1) & capm1;
    }
    slot_of_row[i] = (int)idx;
    atomicMin(&rep[idx], i);                 // deterministic representative
    atomicAdd(&cnt[idx], 1);
    atomicAdd(&sum[idx * 4 + 0], (float)c.x);
    atomicAdd(&sum[idx * 4 + 1], (float)c.y);
    atomicAdd(&sum[idx * 4 + 2], (float)c.z);
    atomicAdd(&sum[idx * 4 + 3], (float)c.w);
}

// ---------------------------------------------------------------------------
// Exclusive prefix sum over "is-representative" flags -> cluster ids
// (first-occurrence ordering; deterministic)
// ---------------------------------------------------------------------------
__global__ void k_scan1(const int* __restrict__ slot_of_row, const int* __restrict__ rep,
                        int* __restrict__ rank, int* __restrict__ blocksums, int N) {
    __shared__ int part[256];
    const int tid  = threadIdx.x;
    const int base = blockIdx.x * 1024 + tid * 4;
    int f[4], s4 = 0;
#pragma unroll
    for (int j = 0; j < 4; ++j) {
        const int i = base + j;
        int fl = 0;
        if (i < N) { const int sl = slot_of_row[i]; fl = (rep[sl] == i) ? 1 : 0; }
        f[j] = fl; s4 += fl;
    }
    part[tid] = s4;
    __syncthreads();
    for (int o = 1; o < 256; o <<= 1) {
        const int v   = part[tid];
        const int add = (tid >= o) ? part[tid - o] : 0;
        __syncthreads();
        part[tid] = v + add;
        __syncthreads();
    }
    int excl = part[tid] - s4;
#pragma unroll
    for (int j = 0; j < 4; ++j) {
        const int i = base + j;
        if (i < N) rank[i] = excl;
        excl += f[j];
    }
    if (tid == 255) blocksums[blockIdx.x] = part[255];
}

__global__ void k_scan2(int* __restrict__ blocksums, int nblk) {
    __shared__ int sh[1024];
    __shared__ int carry;
    const int tid = threadIdx.x;
    if (tid == 0) carry = 0;
    __syncthreads();
    for (int base = 0; base < nblk; base += 1024) {
        const int i = base + tid;
        const int v = (i < nblk) ? blocksums[i] : 0;
        sh[tid] = v;
        __syncthreads();
        for (int o = 1; o < 1024; o <<= 1) {
            const int x   = sh[tid];
            const int add = (tid >= o) ? sh[tid - o] : 0;
            __syncthreads();
            sh[tid] = x + add;
            __syncthreads();
        }
        const int ex = sh[tid] - v + carry;
        __syncthreads();
        if (tid == 0) carry += sh[1023];
        __syncthreads();
        if (i < nblk) blocksums[i] = ex;
    }
}

__global__ void k_scan3(int* __restrict__ rank, const int* __restrict__ blocksums, int N) {
    const int i = blockIdx.x * blockDim.x + threadIdx.x;
    if (i < N) rank[i] += blocksums[i >> 10];
}

__global__ void k_zero_i32(int* __restrict__ p, int n) {
    const int i = blockIdx.x * blockDim.x + threadIdx.x;
    if (i < n) p[i] = 0;
}

__global__ void k_finalize(const int* __restrict__ slot_of_row, const int* __restrict__ rep,
                           const int* __restrict__ cnt, const float* __restrict__ sum,
                           const int* __restrict__ rank, int* __restrict__ out_coords,
                           int* __restrict__ inv, int N) {
    const int i = blockIdx.x * blockDim.x + threadIdx.x;
    if (i >= N) return;
    const int s   = slot_of_row[i];
    const int r   = rep[s];
    const int cid = rank[r];
    inv[i] = cid;
    if (i == r) {                                  // one writer per cluster
        float cn = (float)cnt[s];
        if (cn < 1.0f) cn = 1.0f;
        out_coords[(size_t)cid * 4 + 0] = (int)(sum[s * 4 + 0] / cn);  // trunc == astype(int32)
        out_coords[(size_t)cid * 4 + 1] = (int)(sum[s * 4 + 1] / cn);
        out_coords[(size_t)cid * 4 + 2] = (int)(sum[s * 4 + 2] / cn);
        out_coords[(size_t)cid * 4 + 3] = (int)(sum[s * 4 + 3] / cn);
    }
}

// ---------------------------------------------------------------------------
extern "C" void kernel_launch(void* const* d_in, const int* in_sizes, int n_in,
                              void* d_out, int out_size, void* d_ws, size_t ws_size,
                              hipStream_t stream) {
    const float*         feats  = (const float*)d_in[0];
    const int*           coords = (const int*)d_in[1];
    const unsigned char* mask   = (const unsigned char*)d_in[2];   // jnp bool = 1 byte
    const float* W1 = (const float*)d_in[3];
    const float* b1 = (const float*)d_in[4];
    const float* W2 = (const float*)d_in[5];
    const float* b2 = (const float*)d_in[6];
    const float* W3 = (const float*)d_in[7];
    const float* b3 = (const float*)d_in[8];
    const int N = in_sizes[2];                     // mask has N elements

    // ---- workspace carve-up (256B aligned) --------------------------------
    char*  ws  = (char*)d_ws;
    size_t off = 0;
    auto alloc = [&](size_t bytes) -> char* {
        char* p = ws + off;
        off += (bytes + 255) & ~(size_t)255;
        return p;
    };
    _Float16* W1h         = (_Float16*)alloc(64 * 32 * sizeof(_Float16));
    _Float16* W2h         = (_Float16*)alloc(32 * 16 * sizeof(_Float16));
    _Float16* W3h         = (_Float16*)alloc(16 * 3 * sizeof(_Float16));
    int*      newc        = (int*)alloc((size_t)N * 4 * sizeof(int));
    int*      slot_of_row = (int*)alloc((size_t)N * sizeof(int));
    int*      rank        = (int*)alloc((size_t)N * sizeof(int));
    int*      blocksums   = (int*)alloc(4096 * sizeof(int));

    size_t rem = (ws_size > off) ? (ws_size - off) : 0;
    size_t cap = 1024;                              // 32 B per slot
    while ((cap << 1) * 32 <= rem && cap < ((size_t)1 << 22)) cap <<= 1;
    unsigned long long* keys = (unsigned long long*)alloc(cap * 8);
    int*   rep = (int*)alloc(cap * 4);
    int*   cnt = (int*)alloc(cap * 4);
    float* sum = (float*)alloc(cap * 16);
    const unsigned capm1 = (unsigned)cap - 1u;

    // ---- output layout: [offsets f32 3N | out_coords i32 4N | inv i32 N] --
    float* off_out = (float*)d_out;
    int*   outc    = ((int*)d_out) + (size_t)3 * N;
    int*   inv     = ((int*)d_out) + (size_t)7 * N;

    const int T = 256;
    // 1) weights -> f16
    k_cvt_weights<<<(64 * 32 + T - 1) / T, T, 0, stream>>>(W1, W2, W3, W1h, W2h, W3h);
    // 2) hash table init
    k_init_table<<<((int)cap + T - 1) / T, T, 0, stream>>>(keys, rep, cnt, sum, (int)cap);
    // 3) WMMA decoder (grid-stride over 16-row tiles; 8 waves/block)
    k_decoder<<<1024, 256, 0, stream>>>(feats, coords, mask, W1h, b1, W2h, b2, W3h, b3,
                                        off_out, newc, N);
    // 4) unique via hash insert
    k_hash_insert<<<(N + T - 1) / T, T, 0, stream>>>(newc, keys, rep, cnt, sum,
                                                     slot_of_row, N, capm1);
    // 5) cluster-id ranking: exclusive scan of is-representative flags
    const int nblk = (N + 1023) / 1024;            // <= 4096 supported
    k_scan1<<<nblk, 256, 0, stream>>>(slot_of_row, rep, rank, blocksums, N);
    k_scan2<<<1, 1024, 0, stream>>>(blocksums, nblk);
    k_scan3<<<(N + T - 1) / T, T, 0, stream>>>(rank, blocksums, N);
    // 6) zero-fill out_coords (unique 'fill' rows), then write means + inverse map
    k_zero_i32<<<(4 * N + T - 1) / T, T, 0, stream>>>(outc, 4 * N);
    k_finalize<<<(N + T - 1) / T, T, 0, stream>>>(slot_of_row, rep, cnt, sum, rank,
                                                  outc, inv, N);
}
